// LSTM_23596550324659
// MI455X (gfx1250) — matrix operand order repaired and checked by reading, not compiled, not run on hardware
//
#include <hip/hip_runtime.h>
#include <hip/hip_bf16.h>

// ---------------------------------------------------------------------------
// 2-layer LSTM with L1-normalized gates, B=128, T=512, F=H=512.
// Strategy: bf16 WMMA (v_wmma_f32_16x16x32_bf16) for the per-step fused GEMM
//   gates[t] = [h_{t-1} | x_t] @ [Whh | Wih]^T + (bih + bhh)
// then a per-batch-row pointwise kernel (L1 norm + sigmoid + cell update).
// Weights converted once to bf16 (8 MB total -> resident in 192 MB L2).
// ---------------------------------------------------------------------------

#define Bn   128
#define Tn   512
#define Fn   512
#define Hn   512
#define G4   2048   // 4*H

typedef __attribute__((ext_vector_type(16))) __bf16 v16bf;
typedef __attribute__((ext_vector_type(8)))  __bf16 v8bf;
typedef __attribute__((ext_vector_type(8)))  float  v8f;

// ---------------- fragment loaders (wave32 WMMA, contiguous k order in both A and B) -------------
// A (16x32 bf16, M x K): lane m = lane&15; half = lane>>4.
//   element j holds K = k0 + half*16 + j  -> 16 contiguous bf16 (32B).
__device__ __forceinline__ v16bf load_a_frag(const __bf16* row, int half, int k0) {
    return *(const v16bf*)(row + k0 + half * 16);
}

// B (32x16 bf16, K x N): lane n = lane&15; half = lane>>4.
//   element j holds K = k0 + half*16 + j  -> 16 contiguous bf16 (32B).
__device__ __forceinline__ v16bf load_b_frag(const __bf16* wrow, int half, int k0) {
    return *(const v16bf*)(wrow + k0 + half * 16);
}

// ---------------- per-step fused GEMM kernel -------------------------------
// grid = 128 blocks x 256 threads (8 waves); one wave per 16x16 output tile.
// tiles: mt in [0,8) over B, nt in [0,128) over 4H. K = 512 (h) + 512 (x).
__global__ void lstm_step_gemm(
    const __bf16* __restrict__ hb,     // [B, H]  previous hidden, bf16
    const __bf16* __restrict__ inp,    // input rows for this t (base already offset)
    long                       in_rstride, // elements between batch rows of inp
    const __bf16* __restrict__ Whh,    // [4H, H] bf16 (this layer)
    const __bf16* __restrict__ Wih,    // [4H, F] bf16 (this layer)
    const float*  __restrict__ bias,   // [4H] = bih + bhh
    float*        __restrict__ gates)  // [B, 4H] f32
{
    const int wave = threadIdx.x >> 5;
    const int lane = threadIdx.x & 31;
    const int tile = blockIdx.x * 8 + wave;   // 0..1023
    const int mt   = tile & 7;                // 0..7   (B tiles)
    const int nt   = tile >> 3;               // 0..127 (4H tiles)
    const int half = lane >> 4;
    const int ln16 = lane & 15;

    const __bf16* arow_h = hb  + (long)(mt * 16 + ln16) * Hn;
    const __bf16* arow_x = inp + (long)(mt * 16 + ln16) * in_rstride;
    const __bf16* brow_h = Whh + (long)(nt * 16 + ln16) * Hn;
    const __bf16* brow_x = Wih + (long)(nt * 16 + ln16) * Fn;

    v8f acc = {};
#pragma unroll 4
    for (int k0 = 0; k0 < Hn; k0 += 32) {          // recurrent half
        v16bf a = load_a_frag(arow_h, half, k0);
        v16bf b = load_b_frag(brow_h, half, k0);
        acc = __builtin_amdgcn_wmma_f32_16x16x32_bf16(
                  false, a, false, b, (short)0, acc, false, false);
    }
#pragma unroll 4
    for (int k0 = 0; k0 < Fn; k0 += 32) {          // input half
        v16bf a = load_a_frag(arow_x, half, k0);
        v16bf b = load_b_frag(brow_x, half, k0);
        acc = __builtin_amdgcn_wmma_f32_16x16x32_bf16(
                  false, a, false, b, (short)0, acc, false, false);
    }

    // C/D layout: n = lane&15 ; m = (lane>>4)*8 + r
    const int   gn  = nt * 16 + ln16;
    const float bv  = bias[gn];
    float* out = gates + (long)(mt * 16 + half * 8) * G4 + gn;
#pragma unroll
    for (int r = 0; r < 8; ++r) out[(long)r * G4] = acc[r] + bv;
}

// ---------------- per-step pointwise kernel --------------------------------
// one block per batch row; 512 threads = 16 waves; h = threadIdx.x.
__global__ void lstm_step_point(
    const float* __restrict__ gates,   // [B, 4H]
    float*       __restrict__ c,       // [B, H] cell state (in/out)
    __bf16*      __restrict__ hb,      // [B, H] next hidden bf16 (out)
    float*                    outf,    // f32 hidden out (or null), [b*outf_bs + h]
    long                      outf_bs,
    __bf16*                   out16,   // bf16 hidden out (or null), [b*out16_bs + h]
    long                      out16_bs)
{
    __shared__ float s0[Hn], s1[Hn], s2[Hn];
    const int b = blockIdx.x;
    const int h = threadIdx.x;
    const long base = (long)b * G4;

    const float ig = gates[base + h];
    const float fg = gates[base + Hn + h];
    const float cg = gates[base + 2 * Hn + h];   // NOT normalized / activated
    const float og = gates[base + 3 * Hn + h];

    s0[h] = fabsf(ig); s1[h] = fabsf(fg); s2[h] = fabsf(og);
    __syncthreads();
#pragma unroll
    for (int off = Hn / 2; off > 0; off >>= 1) {
        if (h < off) { s0[h] += s0[h + off]; s1[h] += s1[h + off]; s2[h] += s2[h + off]; }
        __syncthreads();
    }
    const float inv_i = 1.0f / fmaxf(s0[0], 1e-12f);
    const float inv_f = 1.0f / fmaxf(s1[0], 1e-12f);
    const float inv_o = 1.0f / fmaxf(s2[0], 1e-12f);

    const float is = 1.0f / (1.0f + __expf(-ig * inv_i));
    const float fs = 1.0f / (1.0f + __expf(-fg * inv_f));
    const float os = 1.0f / (1.0f + __expf(-og * inv_o));

    const long ch = (long)b * Hn + h;
    const float nc = fs * c[ch] + is * cg;
    const float nh = os * fmaxf(nc, 0.0f);

    c[ch]  = nc;
    hb[ch] = (__bf16)nh;
    if (outf)  outf[(long)b * outf_bs + h]   = nh;
    if (out16) out16[(long)b * out16_bs + h] = (__bf16)nh;
}

// ---------------- small helper kernels -------------------------------------
__global__ void f2bf_kernel(const float* __restrict__ in, __bf16* __restrict__ out, long n) {
    long i = (long)blockIdx.x * blockDim.x + threadIdx.x;
    if (i < n) out[i] = (__bf16)in[i];
}

__global__ void bias_comb_kernel(const float* __restrict__ bih, const float* __restrict__ bhh,
                                 float* __restrict__ out, int n) {
    int i = blockIdx.x * blockDim.x + threadIdx.x;
    if (i < n) out[i] = bih[i] + bhh[i];
}

__global__ void lstm_init_kernel(const float* __restrict__ h0, const float* __restrict__ c0,
                                 int layer, __bf16* __restrict__ hb, float* __restrict__ c) {
    int i = blockIdx.x * blockDim.x + threadIdx.x;   // 0..B*H-1
    int b = i >> 9, h = i & (Hn - 1);
    long src = (long)b * 2 * Hn + (long)layer * Hn + h;
    hb[i] = (__bf16)h0[src];
    c[i]  = c0[src];
}

// ---------------- host launcher --------------------------------------------
extern "C" void kernel_launch(void* const* d_in, const int* in_sizes, int n_in,
                              void* d_out, int out_size, void* d_ws, size_t ws_size,
                              hipStream_t stream) {
    const float* x   = (const float*)d_in[0];   // [B, T, F]
    const float* h0  = (const float*)d_in[1];   // [B, L, H]
    const float* c0  = (const float*)d_in[2];   // [B, L, H]
    const float* Wih = (const float*)d_in[3];   // [L, 4H, F]
    const float* bih = (const float*)d_in[4];   // [L, 4H]
    const float* Whh = (const float*)d_in[5];   // [L, 4H, H]
    const float* bhh = (const float*)d_in[6];   // [L, 4H]
    float* out = (float*)d_out;                 // [B, T, H]

    // workspace layout
    char* p = (char*)d_ws;
    __bf16* xb    = (__bf16*)p;  p += (long)Bn * Tn * Fn * 2;   // 64 MiB
    __bf16* seqb  = (__bf16*)p;  p += (long)Tn * Bn * Hn * 2;   // 64 MiB (layer-0 hidden seq)
    __bf16* Wihb  = (__bf16*)p;  p += (long)2 * G4 * Fn * 2;    // 4 MiB
    __bf16* Whhb  = (__bf16*)p;  p += (long)2 * G4 * Hn * 2;    // 4 MiB
    float*  biasc = (float*)p;   p += (long)2 * G4 * 4;
    float*  gates = (float*)p;   p += (long)Bn * G4 * 4;        // 1 MiB
    float*  cbuf  = (float*)p;   p += (long)Bn * Hn * 4;
    __bf16* hbuf  = (__bf16*)p;  p += (long)Bn * Hn * 2;

    // one-time conversions
    {
        long nx = (long)Bn * Tn * Fn;
        f2bf_kernel<<<(unsigned)((nx + 255) / 256), 256, 0, stream>>>(x, xb, nx);
        long nw = (long)2 * G4 * Fn;
        f2bf_kernel<<<(unsigned)((nw + 255) / 256), 256, 0, stream>>>(Wih, Wihb, nw);
        f2bf_kernel<<<(unsigned)((nw + 255) / 256), 256, 0, stream>>>(Whh, Whhb, nw);
        bias_comb_kernel<<<(2 * G4 + 255) / 256, 256, 0, stream>>>(bih, bhh, biasc, 2 * G4);
    }

    for (int l = 0; l < 2; ++l) {
        lstm_init_kernel<<<(Bn * Hn) / 256, 256, 0, stream>>>(h0, c0, l, hbuf, cbuf);
        const __bf16* WhhL  = Whhb + (long)l * G4 * Hn;
        const __bf16* WihL  = Wihb + (long)l * G4 * Fn;
        const float*  biasL = biasc + (long)l * G4;

        for (int t = 0; t < Tn; ++t) {
            const __bf16* inp; long rstride;
            if (l == 0) { inp = xb   + (long)t * Fn;      rstride = (long)Tn * Fn; }
            else        { inp = seqb + (long)t * Bn * Hn; rstride = Hn; }

            lstm_step_gemm<<<128, 256, 0, stream>>>(hbuf, inp, rstride,
                                                    WhhL, WihL, biasL, gates);

            float*  outf;  long outf_bs;
            __bf16* out16; long out16_bs;
            if (l == 0) { outf = nullptr; outf_bs = 0;
                          out16 = seqb + (long)t * Bn * Hn; out16_bs = Hn; }
            else        { outf = out + (long)t * Hn; outf_bs = (long)Tn * Hn;
                          out16 = nullptr; out16_bs = 0; }

            lstm_step_point<<<Bn, Hn, 0, stream>>>(gates, cbuf, hbuf,
                                                   outf, outf_bs, out16, out16_bs);
        }
    }
}
